// APQBGenerativeModel_59004260712632
// MI455X (gfx1250) — matrix operand order repaired and checked by reading, not compiled, not run on hardware
//
#include <hip/hip_runtime.h>
#include <hip/hip_bf16.h>
#include <math.h>

// ---------------- model constants ----------------
#define B_  2
#define S_  1024
#define D_  1024
#define H_  16
#define L_  3
#define F_  4096
#define V_  32000
#define HD_ 64
#define EPS_ 1e-5f

typedef __attribute__((ext_vector_type(16))) __bf16          v16bf;
typedef __attribute__((ext_vector_type(16))) unsigned short  v16u;
typedef __attribute__((ext_vector_type(8)))  unsigned short  v8u;
typedef __attribute__((ext_vector_type(8)))  float           v8f;

__device__ __forceinline__ unsigned short f2bf(float f) {
  unsigned int u = __float_as_uint(f);
  u += 0x7FFFu + ((u >> 16) & 1u);        // round-to-nearest-even
  return (unsigned short)(u >> 16);
}

// ---------------- elementwise kernels ----------------
__global__ __launch_bounds__(256) void k_f32_to_bf16(const float* __restrict__ in,
                                                     unsigned short* __restrict__ out,
                                                     long long n) {
  long long i = (long long)blockIdx.x * 256 + threadIdx.x;
  if (i < n) out[i] = f2bf(in[i]);
}

__global__ __launch_bounds__(256) void k_embed(const int* __restrict__ x,
                                               const float* __restrict__ tok,
                                               const float* __restrict__ pos,
                                               float* __restrict__ h) {
  long long i = (long long)blockIdx.x * 256 + threadIdx.x;
  long long tot = (long long)B_ * S_ * D_;
  if (i >= tot) return;
  int d = (int)(i % D_);
  long long bs = i / D_;
  int s = (int)(bs % S_);
  int t = x[bs];
  h[i] = tok[(long long)t * D_ + d] + pos[(long long)s * D_ + d];
}

// one block per row; fp32 in -> normalized bf16 out (fused cast for WMMA path)
__global__ __launch_bounds__(256) void k_layernorm_bf16(const float* __restrict__ in,
                                                        const float* __restrict__ g,
                                                        const float* __restrict__ b,
                                                        unsigned short* __restrict__ out) {
  __shared__ float sbuf[256];
  const int tid = threadIdx.x;
  const long long row = blockIdx.x;
  const float* xr = in + row * D_;
  float lsum = 0.f;
  for (int d = tid; d < D_; d += 256) lsum += xr[d];
  sbuf[tid] = lsum; __syncthreads();
  for (int off = 128; off > 0; off >>= 1) {
    if (tid < off) sbuf[tid] += sbuf[tid + off];
    __syncthreads();
  }
  float mu = sbuf[0] * (1.0f / D_);
  __syncthreads();
  float lvar = 0.f;
  for (int d = tid; d < D_; d += 256) { float t = xr[d] - mu; lvar += t * t; }
  sbuf[tid] = lvar; __syncthreads();
  for (int off = 128; off > 0; off >>= 1) {
    if (tid < off) sbuf[tid] += sbuf[tid + off];
    __syncthreads();
  }
  float rstd = rsqrtf(sbuf[0] * (1.0f / D_) + EPS_);
  unsigned short* orow = out + row * D_;
  for (int d = tid; d < D_; d += 256)
    orow[d] = f2bf((xr[d] - mu) * rstd * g[d] + b[d]);
}

// [B,S,H,HD] bf16 -> [B,H,S,HD] bf16 for q,k,v at once
__global__ __launch_bounds__(256) void k_pack_heads(const unsigned short* __restrict__ q,
                                                    const unsigned short* __restrict__ k,
                                                    const unsigned short* __restrict__ v,
                                                    unsigned short* __restrict__ qh,
                                                    unsigned short* __restrict__ kh,
                                                    unsigned short* __restrict__ vh) {
  long long i = (long long)blockIdx.x * 256 + threadIdx.x;
  long long tot = (long long)B_ * S_ * D_;
  if (i >= tot) return;
  int e = (int)(i % HD_);
  long long r = i / HD_;
  int hh = (int)(r % H_);
  long long bs = r / H_;
  int s = (int)(bs % S_);
  int b = (int)(bs / S_);
  long long dst = (((long long)b * H_ + hh) * S_ + s) * HD_ + e;
  qh[dst] = q[i]; kh[dst] = k[i]; vh[dst] = v[i];
}

// [B,H,S,HD] bf16 -> [B,S,H*HD] bf16
__global__ __launch_bounds__(256) void k_merge_heads(const unsigned short* __restrict__ ah,
                                                     unsigned short* __restrict__ am) {
  long long i = (long long)blockIdx.x * 256 + threadIdx.x;
  long long tot = (long long)B_ * S_ * D_;
  if (i >= tot) return;
  int e = (int)(i % HD_);
  long long r = i / HD_;
  int hh = (int)(r % H_);
  long long bs = r / H_;
  int s = (int)(bs % S_);
  int b = (int)(bs / S_);
  am[i] = ah[(((long long)b * H_ + hh) * S_ + s) * HD_ + e];
}

// row softmax over S, fp32 scores -> bf16 probs
__global__ __launch_bounds__(256) void k_softmax_bf16(const float* __restrict__ scores,
                                                      unsigned short* __restrict__ probs) {
  __shared__ float sbuf[256];
  const int tid = threadIdx.x;
  const long long row = blockIdx.x;
  const float* r = scores + row * S_;
  float lmax = -INFINITY;
  for (int j = tid; j < S_; j += 256) lmax = fmaxf(lmax, r[j]);
  sbuf[tid] = lmax; __syncthreads();
  for (int off = 128; off > 0; off >>= 1) {
    if (tid < off) sbuf[tid] = fmaxf(sbuf[tid], sbuf[tid + off]);
    __syncthreads();
  }
  float m = sbuf[0]; __syncthreads();
  float lsum = 0.f;
  for (int j = tid; j < S_; j += 256) lsum += __expf(r[j] - m);
  sbuf[tid] = lsum; __syncthreads();
  for (int off = 128; off > 0; off >>= 1) {
    if (tid < off) sbuf[tid] += sbuf[tid + off];
    __syncthreads();
  }
  float inv = 1.0f / sbuf[0];
  unsigned short* prow = probs + row * S_;
  for (int j = tid; j < S_; j += 256) prow[j] = f2bf(__expf(r[j] - m) * inv);
}

// ---------------- WMMA fragment loaders ----------------
// sA: row-major [row][k], pitch 40.  A 16x32 bf16 layout -> two contiguous 16B runs per lane.
__device__ __forceinline__ v16bf load_frag_a(const unsigned short* sA, int row0, int lane) {
  const int r = lane & 15, half = lane >> 4;
  const unsigned short* base = sA + (row0 + r) * 40 + 8 * half;
  union { struct { v8u lo; v8u hi; } p; v16bf b; } f;
  f.p.lo = *(const v8u*)(base);        // k = 8*half .. +7
  f.p.hi = *(const v8u*)(base + 16);   // k = 16+8*half .. +7
  return f.b;
}

// sB: column-major [n][k], pitch 40.  B 32x16 layout (lane c, same k order as A)
// -> two contiguous 16B runs per lane.
__device__ __forceinline__ v16bf load_frag_b(const unsigned short* sB, int col0, int lane) {
  const int c = lane & 15, half = lane >> 4;
  const unsigned short* base = sB + (col0 + c) * 40 + 8 * half;
  union { struct { v8u lo; v8u hi; } p; v16bf b; } f;
  f.p.lo = *(const v8u*)(base);
  f.p.hi = *(const v8u*)(base + 16);
  return f.b;
}

// ---------------- batched bf16 WMMA GEMM with fused epilogue ----------------
// flags: bit0 = causal mask (col>row -> -inf), bit1 = exact GELU, bit2 = bf16 output
#define FLAG_CAUSAL 1
#define FLAG_GELU   2
#define FLAG_OUTBF  4

__global__ __launch_bounds__(256)
void k_gemm_bf16_wmma(const unsigned short* __restrict__ A, long long sAb, int lda,
                      const unsigned short* __restrict__ Bm, long long sBb, int ldbK, int ldbN,
                      const float* __restrict__ bias,
                      const float* __restrict__ resid, int ldr,
                      float* __restrict__ Cf, unsigned short* __restrict__ Cb,
                      long long sCb, int ldc,
                      int M, int N, int K, float scale, int flags) {
  __shared__ unsigned short sA[128 * 40];   // rows x k   (row-major, pitch 40)
  __shared__ unsigned short sB[128 * 40];   // cols x k   (column-major, pitch 40)

  const int tid  = threadIdx.x;
  const int lane = tid & 31;
  const int wave = tid >> 5;
  const int wm   = wave & 1;   // 2 waves along M  (64 rows each)
  const int wn   = wave >> 1;  // 4 waves along N  (32 cols each)
  const int bM   = blockIdx.y * 128;
  const int bN   = blockIdx.x * 128;
  const int z    = blockIdx.z;

  const unsigned short* Ab = A  + (long long)z * sAb;
  const unsigned short* Bb = Bm + (long long)z * sBb;

  const bool aInterior = (bM + 128 <= M);
  const bool bInterior = (bN + 128 <= N);

  v8f acc[4][2];
  v8f zero = {};
#pragma unroll
  for (int mi = 0; mi < 4; ++mi)
#pragma unroll
    for (int ni = 0; ni < 2; ++ni) acc[mi][ni] = zero;

  for (int kt = 0; kt < K; kt += 32) {
    // ---- stage A tile 128x32 ----
    if (aInterior) {
#pragma unroll
      for (int p2 = 0; p2 < 2; ++p2) {
        int m = (tid >> 2) + p2 * 64;
        int chunk = (tid & 3) * 8;
        *(v8u*)(sA + m * 40 + chunk) =
            *(const v8u*)(Ab + (long long)(bM + m) * lda + kt + chunk);
      }
      if (kt + 32 < K)  // speculative prefetch of next K tile (global_prefetch_b8)
        __builtin_prefetch(Ab + (long long)(bM + (tid >> 1)) * lda + kt + 32, 0, 1);
    } else {
#pragma unroll
      for (int it = 0; it < 16; ++it) {
        int i = tid + it * 256;
        int m = i >> 5, kk = i & 31;
        int gm = bM + m;
        int gmc = gm < M ? gm : (M - 1);                 // clamp: no divergent branch
        unsigned short v = Ab[(long long)gmc * lda + kt + kk];
        sA[m * 40 + kk] = (gm < M) ? v : (unsigned short)0;
      }
    }
    // ---- stage B tile 32x128 into column-major sB[n][k] ----
    if (bInterior) {
      if (ldbN == 1) {
        // rows contiguous along n (weights): vector-load 2 k-rows, packed b32 scatter
        int kk = (tid >> 4) * 2;          // 0..30 even
        int n0 = (tid & 15) * 8;          // 0..120
        const unsigned short* s0 = Bb + (long long)(kt + kk) * ldbK + bN + n0;
        v8u r0 = *(const v8u*)s0;
        v8u r1 = *(const v8u*)(s0 + ldbK);
#pragma unroll
        for (int q = 0; q < 8; ++q) {
          unsigned int pk = (unsigned int)r0[q] | ((unsigned int)r1[q] << 16);
          *(unsigned int*)(sB + (n0 + q) * 40 + kk) = pk;
        }
      } else {
        // k contiguous per column (K^T in attention scores): straight b128 copies
#pragma unroll
        for (int p2 = 0; p2 < 2; ++p2) {
          int n = (tid >> 2) + p2 * 64;
          int chunk = (tid & 3) * 8;
          *(v8u*)(sB + n * 40 + chunk) =
              *(const v8u*)(Bb + (long long)(bN + n) * ldbN + kt + chunk);
        }
      }
    } else {
#pragma unroll
      for (int it = 0; it < 16; ++it) {
        int i = tid + it * 256;
        int n = i >> 5, kk = i & 31;
        int gn = bN + n;
        int gnc = gn < N ? gn : (N - 1);
        unsigned short v = Bb[(long long)(kt + kk) * ldbK + (long long)gnc * ldbN];
        sB[n * 40 + kk] = (gn < N) ? v : (unsigned short)0;
      }
    }
    __syncthreads();

    v16bf bfrag0 = load_frag_b(sB, wn * 32,      lane);
    v16bf bfrag1 = load_frag_b(sB, wn * 32 + 16, lane);
#pragma unroll
    for (int mi = 0; mi < 4; ++mi) {
      v16bf afrag = load_frag_a(sA, wm * 64 + mi * 16, lane);
      acc[mi][0] = __builtin_amdgcn_wmma_f32_16x16x32_bf16(
          false, afrag, false, bfrag0, (short)0, acc[mi][0], false, false);
      acc[mi][1] = __builtin_amdgcn_wmma_f32_16x16x32_bf16(
          false, afrag, false, bfrag1, (short)0, acc[mi][1], false, false);
    }
    __syncthreads();
  }

  // epilogue: C/D layout -> lane 0-15: N=lane, M=r ; lane 16-31: N=lane-16, M=8+r
  const bool full = aInterior && bInterior;
  const int half = lane >> 4;
  const int cn   = lane & 15;
#pragma unroll
  for (int mi = 0; mi < 4; ++mi) {
#pragma unroll
    for (int ni = 0; ni < 2; ++ni) {
#pragma unroll
      for (int r = 0; r < 8; ++r) {
        int row = bM + wm * 64 + mi * 16 + r + 8 * half;
        int col = bN + wn * 32 + ni * 16 + cn;
        if (full || (row < M && col < N)) {
          float v = acc[mi][ni][r] * scale;
          if (bias) v += bias[col];
          if ((flags & FLAG_CAUSAL) && (col > row)) v = -INFINITY;
          if (flags & FLAG_GELU) v = 0.5f * v * (1.0f + erff(v * 0.70710678118654752f));
          if (resid) v += resid[(long long)row * ldr + col];
          long long off = (long long)z * sCb + (long long)row * ldc + col;
          if (flags & FLAG_OUTBF) Cb[off] = f2bf(v);
          else                    Cf[off] = v;
        }
      }
    }
  }
}

// ---------------- host orchestration ----------------
extern "C" void kernel_launch(void* const* d_in, const int* in_sizes, int n_in,
                              void* d_out, int out_size, void* d_ws, size_t ws_size,
                              hipStream_t stream) {
  (void)in_sizes; (void)n_in; (void)out_size; (void)ws_size;

  const int*   x     = (const int*)  d_in[0];
  const float* tok   = (const float*)d_in[1];
  const float* pos   = (const float*)d_in[2];
  const float* Wq    = (const float*)d_in[3];
  const float* bq    = (const float*)d_in[4];
  const float* Wk    = (const float*)d_in[5];
  const float* bk    = (const float*)d_in[6];
  const float* Wv    = (const float*)d_in[7];
  const float* bv    = (const float*)d_in[8];
  const float* Wo    = (const float*)d_in[9];
  const float* bo    = (const float*)d_in[10];
  const float* ln1g  = (const float*)d_in[11];
  const float* ln1b  = (const float*)d_in[12];
  const float* ln2g  = (const float*)d_in[13];
  const float* ln2b  = (const float*)d_in[14];
  const float* W1    = (const float*)d_in[15];
  const float* b1    = (const float*)d_in[16];
  const float* W2    = (const float*)d_in[17];
  const float* b2    = (const float*)d_in[18];
  const float* lnfg  = (const float*)d_in[19];
  const float* lnfb  = (const float*)d_in[20];
  const float* Wout  = (const float*)d_in[21];
  const float* bout  = (const float*)d_in[22];
  float* out = (float*)d_out;

  // bump allocator over d_ws
  char* p = (char*)d_ws;
  auto alloc = [&](size_t bytes) -> char* {
    char* r = p;
    p += (bytes + 255) & ~(size_t)255;
    return r;
  };
  const long long NTOK = (long long)B_ * S_;      // 2048
  const long long NHD  = NTOK * D_;               // 2M

  unsigned short* wq_bf   = (unsigned short*)alloc((size_t)L_ * D_ * D_ * 2);
  unsigned short* wk_bf   = (unsigned short*)alloc((size_t)L_ * D_ * D_ * 2);
  unsigned short* wv_bf   = (unsigned short*)alloc((size_t)L_ * D_ * D_ * 2);
  unsigned short* wo_bf   = (unsigned short*)alloc((size_t)L_ * D_ * D_ * 2);
  unsigned short* w1_bf   = (unsigned short*)alloc((size_t)L_ * D_ * F_ * 2);
  unsigned short* w2_bf   = (unsigned short*)alloc((size_t)L_ * F_ * D_ * 2);
  unsigned short* wout_bf = (unsigned short*)alloc((size_t)D_ * V_ * 2);
  float*          h       = (float*)         alloc((size_t)NHD * 4);
  unsigned short* lnout   = (unsigned short*)alloc((size_t)NHD * 2);
  unsigned short* qlin    = (unsigned short*)alloc((size_t)NHD * 2);
  unsigned short* klin    = (unsigned short*)alloc((size_t)NHD * 2);
  unsigned short* vlin    = (unsigned short*)alloc((size_t)NHD * 2);
  unsigned short* qh      = (unsigned short*)alloc((size_t)NHD * 2);
  unsigned short* kh      = (unsigned short*)alloc((size_t)NHD * 2);
  unsigned short* vh      = (unsigned short*)alloc((size_t)NHD * 2);
  float*          scores  = (float*)         alloc((size_t)B_ * H_ * S_ * S_ * 4);
  unsigned short* probs   = (unsigned short*)alloc((size_t)B_ * H_ * S_ * S_ * 2);
  unsigned short* aoh     = (unsigned short*)alloc((size_t)NHD * 2);
  unsigned short* amrg    = (unsigned short*)alloc((size_t)NHD * 2);
  unsigned short* ffn1    = (unsigned short*)alloc((size_t)NTOK * F_ * 2);

  auto cvt = [&](const float* src, unsigned short* dst, long long n) {
    k_f32_to_bf16<<<(unsigned)((n + 255) / 256), 256, 0, stream>>>(src, dst, n);
  };
  auto gemm = [&](const unsigned short* A, long long sAb, int lda,
                  const unsigned short* Bm, long long sBb, int ldbK, int ldbN,
                  const float* bias, const float* resid, int ldr,
                  float* Cf, unsigned short* Cb, long long sCb, int ldc,
                  int M, int N, int K, int batch, float scale, int flags) {
    dim3 grid((unsigned)((N + 127) / 128), (unsigned)((M + 127) / 128), (unsigned)batch);
    k_gemm_bf16_wmma<<<grid, 256, 0, stream>>>(A, sAb, lda, Bm, sBb, ldbK, ldbN,
                                               bias, resid, ldr, Cf, Cb, sCb, ldc,
                                               M, N, K, scale, flags);
  };

  // weights -> bf16 (each launch; no cross-call state)
  cvt(Wq,   wq_bf,   (long long)L_ * D_ * D_);
  cvt(Wk,   wk_bf,   (long long)L_ * D_ * D_);
  cvt(Wv,   wv_bf,   (long long)L_ * D_ * D_);
  cvt(Wv,   wv_bf,   (long long)L_ * D_ * D_);
  cvt(Wo,   wo_bf,   (long long)L_ * D_ * D_);
  cvt(W1,   w1_bf,   (long long)L_ * D_ * F_);
  cvt(W2,   w2_bf,   (long long)L_ * F_ * D_);
  cvt(Wout, wout_bf, (long long)D_ * V_);

  // embedding
  k_embed<<<(unsigned)((NHD + 255) / 256), 256, 0, stream>>>(x, tok, pos, h);

  const int M0 = (int)NTOK;                 // 2048 token rows
  const float attn_scale = 0.125f;          // HD^-0.5 = 1/8

  for (int i = 0; i < L_; ++i) {
    const size_t wDD = (size_t)i * D_ * D_;
    const size_t wDF = (size_t)i * D_ * F_;
    // --- attention ---
    k_layernorm_bf16<<<M0, 256, 0, stream>>>(h, ln1g + i * D_, ln1b + i * D_, lnout);
    gemm(lnout, 0, D_, wq_bf + wDD, 0, D_, 1, bq + i * D_, nullptr, 0,
         nullptr, qlin, 0, D_, M0, D_, D_, 1, 1.f, FLAG_OUTBF);
    gemm(lnout, 0, D_, wk_bf + wDD, 0, D_, 1, bk + i * D_, nullptr, 0,
         nullptr, klin, 0, D_, M0, D_, D_, 1, 1.f, FLAG_OUTBF);
    gemm(lnout, 0, D_, wv_bf + wDD, 0, D_, 1, bv + i * D_, nullptr, 0,
         nullptr, vlin, 0, D_, M0, D_, D_, 1, 1.f, FLAG_OUTBF);
    k_pack_heads<<<(unsigned)((NHD + 255) / 256), 256, 0, stream>>>(qlin, klin, vlin, qh, kh, vh);
    // scores = scale * Q K^T with causal mask, per (b,h); K^T via ldbK=1/ldbN=HD
    gemm(qh, (long long)S_ * HD_, HD_, kh, (long long)S_ * HD_, 1, HD_,
         nullptr, nullptr, 0, scores, nullptr, (long long)S_ * S_, S_,
         S_, S_, HD_, B_ * H_, attn_scale, FLAG_CAUSAL);
    k_softmax_bf16<<<(unsigned)(B_ * H_ * S_), 256, 0, stream>>>(scores, probs);
    // out_h = P V, per (b,h)
    gemm(probs, (long long)S_ * S_, S_, vh, (long long)S_ * HD_, HD_, 1,
         nullptr, nullptr, 0, nullptr, aoh, (long long)S_ * HD_, HD_,
         S_, HD_, S_, B_ * H_, 1.f, FLAG_OUTBF);
    k_merge_heads<<<(unsigned)((NHD + 255) / 256), 256, 0, stream>>>(aoh, amrg);
    // h = h + out @ Wo + bo
    gemm(amrg, 0, D_, wo_bf + wDD, 0, D_, 1, bo + i * D_, h, D_,
         h, nullptr, 0, D_, M0, D_, D_, 1, 1.f, 0);
    // --- FFN ---
    k_layernorm_bf16<<<M0, 256, 0, stream>>>(h, ln2g + i * D_, ln2b + i * D_, lnout);
    gemm(lnout, 0, D_, w1_bf + wDF, 0, F_, 1, b1 + i * F_, nullptr, 0,
         nullptr, ffn1, 0, F_, M0, F_, D_, 1, 1.f, FLAG_GELU | FLAG_OUTBF);
    gemm(ffn1, 0, F_, w2_bf + wDF, 0, D_, 1, b2 + i * D_, h, D_,
         h, nullptr, 0, D_, M0, D_, F_, 1, 1.f, 0);
  }

  // final LN + vocab projection
  k_layernorm_bf16<<<M0, 256, 0, stream>>>(h, lnfg, lnfb, lnout);
  gemm(lnout, 0, D_, wout_bf, 0, V_, 1, bout, nullptr, 0,
       out, nullptr, 0, V_, M0, V_, D_, 1, 1.f, 0);
}
